// ProteinGraphModel_37804302139934
// MI455X (gfx1250) — matrix operand order repaired and checked by reading, not compiled, not run on hardware
//
#include <hip/hip_runtime.h>

// ---------- CDNA5 WMMA types ----------
typedef __attribute__((ext_vector_type(16))) __bf16 v16bf;
typedef __attribute__((ext_vector_type(8)))  float  v8f;

#define RNUM 6

__device__ __forceinline__ unsigned short f2bf(float f) {
  unsigned u = __float_as_uint(f);
  u += 0x7FFFu + ((u >> 16) & 1u);       // round to nearest even
  return (unsigned short)(u >> 16);
}

union FragU { uint4 u[2]; v16bf v; };

// A fragment (16x32 bf16): lane<16 holds M=lane, K = {k0..k0+7, k0+16..k0+23};
// lane>=16 holds M=lane-16, K = {k0+8..k0+15, k0+24..k0+31}. Caller pre-adds the
// (lane&16 ? 8 : 0) element offset; the two 16B chunks are 16 elements apart.
__device__ __forceinline__ v16bf load_a_frag(const unsigned short* p) {
  FragU f;
  f.u[0] = *(const uint4*)(p);
  f.u[1] = *(const uint4*)(p + 16);
  return f.v;
}
// B fragment (32x16 bf16, stored as B^T rows): lane<16 holds N=lane, K={k0..k0+7, k0+16..k0+23};
// lane>=16 holds N=lane-16, K={k0+8..k0+15, k0+24..k0+31}. Caller pre-adds (lane&16 ? 8 : 0);
// the two 16B chunks are 16 elements apart.
__device__ __forceinline__ v16bf load_b_frag(const unsigned short* p) {
  FragU f;
  f.u[0] = *(const uint4*)(p);
  f.u[1] = *(const uint4*)(p + 16);
  return f.v;
}

// ---------- utility kernels ----------
__global__ void k_zero(uint4* __restrict__ p, unsigned long long n4) {
  unsigned long long i = (unsigned long long)blockIdx.x * blockDim.x + threadIdx.x;
  if (i < n4) { uint4 z; z.x = z.y = z.z = z.w = 0u; p[i] = z; }
}

__global__ void k_count(const int* __restrict__ ei, const int* __restrict__ et,
                        unsigned* __restrict__ cnt, int E) {
  int e = blockIdx.x * blockDim.x + threadIdx.x;
  if (e < E) atomicAdd(&cnt[(long)ei[E + e] * RNUM + et[e]], 1u);
}

__global__ void k_inv(const unsigned* __restrict__ cnt, float* __restrict__ inv, long n) {
  long i = (long)blockIdx.x * blockDim.x + threadIdx.x;
  if (i < n) { unsigned c = cnt[i]; inv[i] = 1.0f / (float)(c ? c : 1u); }
}

// scatter-add source features into (dst, relation) segment sums, float4 at a time
__global__ void k_scatter(const float* __restrict__ h, const int* __restrict__ ei,
                          const int* __restrict__ et, float* __restrict__ seg,
                          int E, int din) {
  int q = din >> 2;
  long idx = (long)blockIdx.x * blockDim.x + threadIdx.x;
  if (idx >= (long)E * q) return;
  int e = (int)(idx / q);
  int i = (int)(idx - (long)e * q) << 2;
  int s = ei[e], d = ei[E + e], r = et[e];
  float4 v = *(const float4*)(h + (size_t)s * din + i);
  float* sp = seg + ((size_t)d * RNUM + r) * din + i;
  atomicAdd(sp + 0, v.x); atomicAdd(sp + 1, v.y);
  atomicAdd(sp + 2, v.z); atomicAdd(sp + 3, v.w);
}

// build bf16 A = [mean_r0 .. mean_r5, h] padded to Kpad, 4 elems / thread
__global__ void k_build_a(const float* __restrict__ h, const float* __restrict__ seg,
                          const float* __restrict__ inv, unsigned short* __restrict__ A,
                          int Nn, int din, int Kpad) {
  int K  = (RNUM + 1) * din;
  int q  = Kpad >> 2;
  long idx = (long)blockIdx.x * blockDim.x + threadIdx.x;
  if (idx >= (long)Nn * q) return;
  int n  = (int)(idx / q);
  int k0 = (int)(idx - (long)n * q) << 2;
  alignas(8) unsigned short out[4];
#pragma unroll
  for (int j = 0; j < 4; ++j) {
    int k = k0 + j; float v;
    if (k < RNUM * din) {
      int r = k / din, i = k - r * din;
      long sgi = (long)n * RNUM + r;
      v = seg[sgi * din + i] * inv[sgi];
    } else if (k < K) {
      v = h[(size_t)n * din + (k - RNUM * din)];
    } else v = 0.0f;
    out[j] = f2bf(v);
  }
  *(uint2*)(A + (size_t)n * Kpad + k0) = *(const uint2*)out;
}

// build bf16 W^T [dout, Kpad]: rows of W^T are columns of [W0;..;W5; root]
__global__ void k_build_w(const float* __restrict__ W, const float* __restrict__ root,
                          unsigned short* __restrict__ WT, int din, int dout, int Kpad) {
  int K = (RNUM + 1) * din;
  long idx = (long)blockIdx.x * blockDim.x + threadIdx.x;
  if (idx >= (long)dout * Kpad) return;
  int o = (int)(idx / Kpad);
  int k = (int)(idx - (long)o * Kpad);
  float v = 0.0f;
  if (k < RNUM * din)      v = W[(size_t)k * dout + o];                 // (r*din+i)*dout+o
  else if (k < K)          v = root[(size_t)(k - RNUM * din) * dout + o];
  WT[(size_t)o * Kpad + k] = f2bf(v);
}

// ---------- bf16 WMMA GEMM: C[Nn,dout] = relu(A[Nn,Kpad] * WT^T + bias) ----------
// One wave computes a 16x64 strip (4 WMMA tiles, A fragment reused 4x).
__global__ void __launch_bounds__(256) k_gemm(const unsigned short* __restrict__ A,
                                              const unsigned short* __restrict__ B,
                                              const float* __restrict__ bias,
                                              float* __restrict__ C,
                                              int Nn, int Kpad, int dout) {
  int wave = blockIdx.x * (blockDim.x >> 5) + (threadIdx.x >> 5);
  int lane = threadIdx.x & 31;
  int groups = dout >> 6;                 // 64 output cols per wave
  int mt = wave / groups;
  if (mt * 16 >= Nn) return;              // uniform per-wave exit
  int g  = wave - mt * groups;
  int m0 = mt << 4;
  int n0 = g << 6;
  int ra = m0 + (lane & 15); if (ra >= Nn) ra = Nn - 1;   // clamp (stores predicated)
  const unsigned short* pA = A + (size_t)ra * Kpad + ((lane & 16) ? 8 : 0);
  const unsigned short* pB = B + (size_t)(n0 + (lane & 15)) * Kpad + ((lane & 16) ? 8 : 0);
  const size_t bstr = (size_t)16 * Kpad;

  v8f c0 = {}, c1 = {}, c2 = {}, c3 = {};
  for (int k0 = 0; k0 < Kpad; k0 += 32) {
    v16bf a  = load_a_frag(pA + k0);
    v16bf b0 = load_b_frag(pB + k0);
    v16bf b1 = load_b_frag(pB + bstr + k0);
    v16bf b2 = load_b_frag(pB + 2 * bstr + k0);
    v16bf b3 = load_b_frag(pB + 3 * bstr + k0);
    c0 = __builtin_amdgcn_wmma_f32_16x16x32_bf16(false, a, false, b0, (short)0, c0, false, false);
    c1 = __builtin_amdgcn_wmma_f32_16x16x32_bf16(false, a, false, b1, (short)0, c1, false, false);
    c2 = __builtin_amdgcn_wmma_f32_16x16x32_bf16(false, a, false, b2, (short)0, c2, false, false);
    c3 = __builtin_amdgcn_wmma_f32_16x16x32_bf16(false, a, false, b3, (short)0, c3, false, false);
  }

  // C/D layout: element r of v8f in lane l -> row m0 + r + 8*(l/16), col n0 + (l%16)
  int rbase = m0 + ((lane >> 4) << 3);
  int col   = n0 + (lane & 15);
#define STORE_TILE(cv, t)                                                     \
  {                                                                           \
    int cc = col + (t) * 16;                                                  \
    float bv = bias[cc];                                                      \
    _Pragma("unroll") for (int r = 0; r < 8; ++r) {                           \
      int row = rbase + r;                                                    \
      if (row < Nn) C[(size_t)row * dout + cc] = fmaxf((cv)[r] + bv, 0.0f);   \
    }                                                                         \
  }
  STORE_TILE(c0, 0) STORE_TILE(c1, 1) STORE_TILE(c2, 2) STORE_TILE(c3, 3)
#undef STORE_TILE
}

// ---------- epilogue: column max (values >= 0 after ReLU -> int-bit atomicMax) ----------
__global__ void k_colmax(const float* __restrict__ h, int* __restrict__ mx, int Nn, int dout) {
  int o = threadIdx.x;
  float m = 0.0f;
  for (int nrow = blockIdx.x; nrow < Nn; nrow += gridDim.x)
    m = fmaxf(m, h[(size_t)nrow * dout + o]);
  atomicMax(&mx[o], __float_as_int(m));
}

// tanh(max) then L2-normalize (tanh monotonic => max(tanh(h)) == tanh(max(h)))
__global__ void k_final(const int* __restrict__ mx, float* __restrict__ out, int dout) {
  __shared__ float s[512];
  int o = threadIdx.x;
  float m = tanhf(__int_as_float(mx[o]));
  s[o] = m * m;
  __syncthreads();
  for (int st = dout >> 1; st > 0; st >>= 1) {
    if (o < st) s[o] += s[o + st];
    __syncthreads();
  }
  out[o] = m * rsqrtf(s[0]);
}

// ---------- host ----------
extern "C" void kernel_launch(void* const* d_in, const int* in_sizes, int n_in,
                              void* d_out, int out_size, void* d_ws, size_t ws_size,
                              hipStream_t stream) {
  const float* x  = (const float*)d_in[0];
  const int*   ei = (const int*)d_in[1];     // [2,E] flat: src then dst
  const int*   et = (const int*)d_in[2];
  const float* Wr[3]   = {(const float*)d_in[3], (const float*)d_in[6], (const float*)d_in[9]};
  const float* rootp[3]= {(const float*)d_in[4], (const float*)d_in[7], (const float*)d_in[10]};
  const float* bias[3] = {(const float*)d_in[5], (const float*)d_in[8], (const float*)d_in[11]};
  const int DIN[3]  = {20, 256, 256};
  const int DOUT[3] = {256, 256, 512};

  const int  Nn = in_sizes[0] / 20;
  const int  E  = in_sizes[2];
  const long NR = (long)Nn * RNUM;

  char* wsp = (char*)d_ws; size_t off = 0;
  auto alloc = [&](size_t bytes) -> void* {
    void* p = wsp + off;
    off += (bytes + 255) & ~(size_t)255;
    return p;
  };
  unsigned*       cnt  = (unsigned*)alloc((size_t)NR * 4);
  float*          inv  = (float*)alloc((size_t)NR * 4);
  float*          seg  = (float*)alloc((size_t)NR * 256 * 4);
  unsigned short* Abf  = (unsigned short*)alloc((size_t)Nn * 1792 * 2);
  unsigned short* WT   = (unsigned short*)alloc((size_t)512 * 1792 * 2);
  float*          hbuf = (float*)alloc((size_t)Nn * 512 * 4);
  int*            mx   = (int*)alloc(512 * 4);
  (void)n_in; (void)out_size; (void)ws_size;

  const int TB = 256;
  // edge counts (fixed across layers)
  {
    unsigned long long n4 = (unsigned long long)(NR / 4);
    k_zero<<<(unsigned)((n4 + TB - 1) / TB), TB, 0, stream>>>((uint4*)cnt, n4);
    k_count<<<(E + TB - 1) / TB, TB, 0, stream>>>(ei, et, cnt, E);
    k_inv<<<(unsigned)((NR + TB - 1) / TB), TB, 0, stream>>>(cnt, inv, NR);
  }

  const float* hprev = x;
  for (int l = 0; l < 3; ++l) {
    int din = DIN[l], dout = DOUT[l];
    int K = (RNUM + 1) * din;
    int Kpad = (K + 31) & ~31;

    unsigned long long segn4 = (unsigned long long)(NR * din / 4);
    k_zero<<<(unsigned)((segn4 + TB - 1) / TB), TB, 0, stream>>>((uint4*)seg, segn4);

    long sc = (long)E * (din / 4);
    k_scatter<<<(unsigned)((sc + TB - 1) / TB), TB, 0, stream>>>(hprev, ei, et, seg, E, din);

    long at = (long)Nn * (Kpad / 4);
    k_build_a<<<(unsigned)((at + TB - 1) / TB), TB, 0, stream>>>(hprev, seg, inv, Abf, Nn, din, Kpad);

    long wt = (long)dout * Kpad;
    k_build_w<<<(unsigned)((wt + TB - 1) / TB), TB, 0, stream>>>(Wr[l], rootp[l], WT, din, dout, Kpad);

    int  mtiles = (Nn + 15) / 16;
    long waves  = (long)mtiles * (dout / 64);
    int  blocks = (int)((waves * 32 + TB - 1) / TB);
    k_gemm<<<blocks, TB, 0, stream>>>(Abf, WT, bias[l], hbuf, Nn, Kpad, dout);

    hprev = hbuf;
  }

  k_zero<<<1, TB, 0, stream>>>((uint4*)mx, 512 / 4);
  k_colmax<<<256, 512, 0, stream>>>(hbuf, mx, Nn, 512);
  k_final<<<1, 512, 0, stream>>>(mx, (float*)d_out, 512);
}
